// Attention_20366734918152
// MI455X (gfx1250) — compile-verified, matrix operand order checked by reading, not run on hardware
//
#include <hip/hip_runtime.h>
#include <hip/hip_bf16.h>

#define B_    4
#define T_    2048
#define E_    512
#define H_    8
#define HD_   64
#define QKW_  8192   /* 2*h*e */
#define HE_   4096   /* h*e   */
#define SCALE_ 0.125f /* 64^-0.5 */

typedef __attribute__((ext_vector_type(16))) __bf16 v16bf;
typedef __attribute__((ext_vector_type(8)))  __bf16 v8bf;
typedef __attribute__((ext_vector_type(8)))  float  v8f;

#if defined(__has_builtin)
#if __has_builtin(__builtin_amdgcn_tensor_load_to_lds) && \
    __has_builtin(__builtin_amdgcn_s_wait_tensorcnt)
#define USE_TDM 1
#endif
#endif

#ifdef USE_TDM
typedef __attribute__((ext_vector_type(4))) unsigned int u32x4;
typedef __attribute__((ext_vector_type(8))) int i32x8;
typedef __attribute__((ext_vector_type(4))) int i32x4;
#endif

static __device__ __forceinline__ __bf16 f2bf(float f) {
  unsigned u = __builtin_bit_cast(unsigned, f);
  unsigned r = (u + 0x7FFFu + ((u >> 16) & 1u)) >> 16;
  unsigned short s = (unsigned short)r;
  return __builtin_bit_cast(__bf16, s);
}

union V16U { v16bf v; v8bf h[2]; };

// One WMMA A/B-operand fragment: lane half h holds K=[kBase+8h,+8) and [kBase+16+8h,+8)
static __device__ __forceinline__ v16bf load_frag(const __bf16* row, int kBase, int half) {
  V16U u;
  u.h[0] = *(const v8bf*)(row + kBase + half * 8);
  u.h[1] = *(const v8bf*)(row + kBase + 16 + half * 8);
  return u.v;
}

static __device__ __forceinline__ v8f wmma_bf16(v16bf a, v16bf b, v8f c) {
  return __builtin_amdgcn_wmma_f32_16x16x32_bf16(false, a, false, b, (short)0, c,
                                                 false, false);
}

// ---------------- prep kernels ----------------
__global__ void k_f32_to_bf16(const float* __restrict__ s, __bf16* __restrict__ d, long n) {
  long i = (long)blockIdx.x * blockDim.x + threadIdx.x;
  if (i < n) d[i] = f2bf(s[i]);
}

__global__ void k_transpose_to_bf16(const float* __restrict__ s, __bf16* __restrict__ d,
                                    int rows, int cols) {
  int i = blockIdx.x * blockDim.x + threadIdx.x;
  if (i < rows * cols) {
    int r = i / cols, c = i % cols;
    d[(long)c * rows + r] = f2bf(s[i]);
  }
}

// ---------------- generic bf16 WMMA GEMM: out = A[M,K] * B[N,K]^T (+bias) ----------------
__global__ void __launch_bounds__(256)
k_gemm_bf16(const __bf16* __restrict__ A, const __bf16* __restrict__ Bm,
            const float* __restrict__ bias, void* __restrict__ out,
            int M, int N, int K,
            int rowsPerBatch, long batchStride, long rowStride, long colStride,
            int storeF32) {
  int wave = threadIdx.x >> 5;
  int lane = threadIdx.x & 31;
  int half = lane >> 4;
  int r    = lane & 15;
  int tilesN = N >> 6;
  int tile = blockIdx.x * (blockDim.x >> 5) + wave;
  int tM = tile / tilesN, tN = tile % tilesN;
  if (tM * 16 >= M) return;

  const __bf16* Arow = A  + (long)(tM * 16 + r) * K;
  const __bf16* Brow = Bm + (long)(tN * 64 + r) * K;

  v8f acc[4] = {};
  for (int k = 0; k < K; k += 32) {
    v16bf a  = load_frag(Arow, k, half);
    v16bf b0 = load_frag(Brow,            k, half);
    v16bf b1 = load_frag(Brow + 16L * K,  k, half);
    v16bf b2 = load_frag(Brow + 32L * K,  k, half);
    v16bf b3 = load_frag(Brow + 48L * K,  k, half);
    acc[0] = wmma_bf16(a, b0, acc[0]);
    acc[1] = wmma_bf16(a, b1, acc[1]);
    acc[2] = wmma_bf16(a, b2, acc[2]);
    acc[3] = wmma_bf16(a, b3, acc[3]);
  }

#pragma unroll
  for (int j = 0; j < 4; ++j) {
#pragma unroll
    for (int v = 0; v < 8; ++v) {
      int m = tM * 16 + v + 8 * half;
      int n = tN * 64 + j * 16 + r;
      float val = acc[j][v] + (bias ? bias[n] : 0.0f);
      long idx = (long)(m / rowsPerBatch) * batchStride +
                 (long)(m % rowsPerBatch) * rowStride + (long)n * colStride;
      if (storeF32) ((float*)out)[idx] = val;
      else          ((__bf16*)out)[idx] = f2bf(val);
    }
  }
}

// ---------------- flash-style log-sparse attention ----------------
// Block = 4 waves sharing one (b,h); wave w owns q rows [qt, qt+16).
// Q tile cached in 128 VGPRs; 32x512 bf16 K tile staged once per block in LDS
// via TDM tensor_load_to_lds (cooperative b128 loads otherwise).
__global__ void __launch_bounds__(128)
k_attention(const __bf16* __restrict__ qk, const __bf16* __restrict__ vT,
            const unsigned char* __restrict__ mask, __bf16* __restrict__ ctx) {
  __shared__ __align__(16) __bf16 ldsK[32 * E_];       // 32 KB staged key tile
  __shared__ __align__(16) __bf16 ldsP[4][16 * 32];    // 4 KB P transpose staging

  int wave = threadIdx.x >> 5, lane = threadIdx.x & 31;
  int half = lane >> 4, r = lane & 15;
  int b = blockIdx.z, h = blockIdx.y;
  int qt = blockIdx.x * 64 + wave * 16;

  const __bf16* Qrow     = qk + ((long)(b * T_ + qt + r) * QKW_ + h * E_);
  const __bf16* KbaseRow = qk + ((long)b * T_ * QKW_ + HE_ + h * E_);
  const __bf16* Vbase    = vT + ((long)b * E_ * T_ + (long)h * HD_ * T_);

  // Cache the wave's entire 16x512 Q tile in registers: 16 fragments = 128 VGPRs.
  v16bf qf[16];
#pragma unroll
  for (int c = 0; c < 16; ++c) qf[c] = load_frag(Qrow, c * 32, half);

  float mcur[8], lcur[8];
  v8f o[4] = {};
#pragma unroll
  for (int v = 0; v < 8; ++v) { mcur[v] = -__builtin_inff(); lcur[v] = 0.f; }

  // Block-uniform key range (max over the 4 waves' causal extents).
  int ktEnd = ((blockIdx.x * 64 + 48 + 16 + 31) >> 5) << 5;
  if (ktEnd > T_) ktEnd = T_;

  for (int kt = 0; kt < ktEnd; kt += 32) {
    // --- per-wave mask bytes; block-level skip of fully-masked tiles ---
    unsigned char mk0[8], mk1[8];
    int anyLocal = 0;
#pragma unroll
    for (int v = 0; v < 8; ++v) {
      int q = qt + v + 8 * half;
      const unsigned char* mp = mask + (long)q * T_ + kt;
      mk0[v] = mp[r];
      mk1[v] = mp[16 + r];
      anyLocal |= (mk0[v] | mk1[v]);
    }
    int anyBlock = __syncthreads_or(anyLocal);   // also fences ldsK reuse
    if (!anyBlock) continue;

    // --- stage 32x512 bf16 K tile into LDS, shared by all 4 waves ---
#ifdef USE_TDM
    if (wave == 0) {
      unsigned long long ga = (unsigned long long)(KbaseRow + (long)kt * QKW_);
      unsigned ldsAddr = (unsigned)(unsigned long long)&ldsK[0];
      // D# group0: count=1 | lds_addr | global_addr[56:0] | type=2
      u32x4 g0 = { 1u, ldsAddr, (unsigned)ga,
                   (unsigned)((ga >> 32) & 0x1FFFFFFu) | 0x80000000u };
      // D# group1: data_size=2B; tensor_dim0=512, tensor_dim1=32;
      //            tile_dim0=512, tile_dim1=32; tensor_dim0_stride=8192
      i32x8 g1 = { (int)0x00010000u,
                   (int)(512u << 16),      // tensor_dim0 lo16 (bits 63:48)
                   (int)(32u  << 16),      // tensor_dim0 hi=0 | tensor_dim1 lo16
                   (int)(512u << 16),      // tensor_dim1 hi=0 | tile_dim0
                   32,                     // tile_dim1=32, tile_dim2=0
                   QKW_,                   // tensor_dim0_stride lo32
                   0, 0 };
      i32x4 gz4 = { 0, 0, 0, 0 };
      i32x8 gz8 = { 0, 0, 0, 0, 0, 0, 0, 0 };
      __builtin_amdgcn_tensor_load_to_lds(g0, g1, gz4, gz4, gz8, 0);
      __builtin_amdgcn_s_wait_tensorcnt(0);
    }
#else
    {
      // 32 rows x 1KB = 2048 x 16B chunks, 16 per thread
      int t = threadIdx.x;
      const __bf16* src = KbaseRow + (long)kt * QKW_;
#pragma unroll
      for (int i = 0; i < 16; ++i) {
        int c = t + i * 128;
        int row = c >> 6, col = (c & 63) * 8;
        *(v8bf*)(&ldsK[row * E_ + col]) = *(const v8bf*)(src + (long)row * QKW_ + col);
      }
    }
#endif
    __syncthreads();

    if (__any(anyLocal)) {
      // --- S = Q K^T over the full 512-wide embedding (K operands from LDS) ---
      v8f s0 = {}, s1 = {};
      const __bf16* K0 = &ldsK[r * E_];
      const __bf16* K1 = &ldsK[(16 + r) * E_];
#pragma unroll
      for (int c = 0; c < 16; ++c) {
        s0 = wmma_bf16(qf[c], load_frag(K0, c * 32, half), s0);
        s1 = wmma_bf16(qf[c], load_frag(K1, c * 32, half), s1);
      }

      // --- online softmax ---
#pragma unroll
      for (int v = 0; v < 8; ++v) {
        float x0 = mk0[v] ? s0[v] * SCALE_ : -__builtin_inff();
        float x1 = mk1[v] ? s1[v] * SCALE_ : -__builtin_inff();
        float t = fmaxf(x0, x1);
#pragma unroll
        for (int off = 1; off < 16; off <<= 1) t = fmaxf(t, __shfl_xor(t, off, 16));
        float mold = mcur[v];
        float mnew = fmaxf(mold, t);
        float alpha = 1.f, p0 = 0.f, p1 = 0.f;
        if (mnew > -1e30f) {
          alpha = __expf(mold - mnew);
          p0 = (x0 > -1e30f) ? __expf(x0 - mnew) : 0.f;
          p1 = (x1 > -1e30f) ? __expf(x1 - mnew) : 0.f;
        }
        float rs = p0 + p1;
#pragma unroll
        for (int off = 1; off < 16; off <<= 1) rs += __shfl_xor(rs, off, 16);
        mcur[v] = mnew;
        lcur[v] = lcur[v] * alpha + rs;
#pragma unroll
        for (int j = 0; j < 4; ++j) o[j][v] *= alpha;
        int m = v + 8 * half;
        ldsP[wave][m * 32 + r]      = f2bf(p0);
        ldsP[wave][m * 32 + 16 + r] = f2bf(p1);
      }

      // --- O += P V (P via LDS transpose; V streamed from global/L2) ---
      const __bf16* Prow = &ldsP[wave][r * 32];
      v16bf aP = load_frag(Prow, 0, half);
#pragma unroll
      for (int j = 0; j < 4; ++j) {
        v16bf bV = load_frag(Vbase + (long)(j * 16 + r) * T_ + kt, 0, half);
        o[j] = wmma_bf16(aP, bV, o[j]);
      }
    }
  }

  // --- normalize and write ctx[b, q, h*64 + d] in bf16 ---
#pragma unroll
  for (int v = 0; v < 8; ++v) {
    int q = qt + v + 8 * half;
    float inv = lcur[v] > 0.f ? 1.0f / lcur[v] : 0.f;
#pragma unroll
    for (int j = 0; j < 4; ++j) {
      ctx[(long)(b * T_ + q) * E_ + h * HD_ + j * 16 + r] = f2bf(o[j][v] * inv);
    }
  }
}

// ---------------- launcher ----------------
extern "C" void kernel_launch(void* const* d_in, const int* in_sizes, int n_in,
                              void* d_out, int out_size, void* d_ws, size_t ws_size,
                              hipStream_t stream) {
  const float* x      = (const float*)d_in[0];
  const float* w_qk   = (const float*)d_in[1];
  const float* w_v    = (const float*)d_in[2];
  const float* b_v    = (const float*)d_in[3];
  const float* w_proj = (const float*)d_in[4];
  const float* b_proj = (const float*)d_in[5];
  const unsigned char* mask = (const unsigned char*)d_in[6];
  float* out = (float*)d_out;

  char* ws = (char*)d_ws;
  __bf16* x_bf   = (__bf16*)ws; ws += (long)B_ * T_ * E_ * 2;
  __bf16* wqk_bf = (__bf16*)ws; ws += (long)QKW_ * E_ * 2;
  __bf16* wvT_bf = (__bf16*)ws; ws += (long)E_ * E_ * 2;
  __bf16* wpT_bf = (__bf16*)ws; ws += (long)E_ * E_ * 2;
  __bf16* qk_bf  = (__bf16*)ws; ws += (long)B_ * T_ * QKW_ * 2;
  __bf16* vT_bf  = (__bf16*)ws; ws += (long)B_ * E_ * T_ * 2;
  __bf16* ctx_bf = (__bf16*)ws; ws += (long)B_ * T_ * E_ * 2;

  {
    long n = (long)B_ * T_ * E_;
    k_f32_to_bf16<<<(int)((n + 255) / 256), 256, 0, stream>>>(x, x_bf, n);
  }
  {
    long n = (long)QKW_ * E_;
    k_f32_to_bf16<<<(int)((n + 255) / 256), 256, 0, stream>>>(w_qk, wqk_bf, n);
  }
  k_transpose_to_bf16<<<(E_ * E_ + 255) / 256, 256, 0, stream>>>(w_v, wvT_bf, E_, E_);
  k_transpose_to_bf16<<<(E_ * E_ + 255) / 256, 256, 0, stream>>>(w_proj, wpT_bf, E_, E_);

  {
    int M = B_ * T_, N = QKW_, K = E_;
    int tiles = (M / 16) * (N / 64);
    k_gemm_bf16<<<tiles / 8, 256, 0, stream>>>(x_bf, wqk_bf, nullptr, qk_bf,
                                               M, N, K, M, 0L, (long)QKW_, 1L, 0);
  }
  {
    int M = B_ * T_, N = E_, K = E_;
    int tiles = (M / 16) * (N / 64);
    k_gemm_bf16<<<tiles / 8, 256, 0, stream>>>(x_bf, wvT_bf, b_v, vT_bf,
                                               M, N, K, T_, (long)E_ * T_, 1L, (long)T_, 0);
  }
  k_attention<<<dim3(T_ / 64, H_, B_), 128, 0, stream>>>(qk_bf, vT_bf, mask, ctx_bf);
  {
    int M = B_ * T_, N = E_, K = E_;
    int tiles = (M / 16) * (N / 64);
    k_gemm_bf16<<<tiles / 8, 256, 0, stream>>>(ctx_bf, wpT_bf, b_proj, out,
                                               M, N, K, M, 0L, (long)E_, 1L, 1);
  }
}
